// FullDualColumn_38156489458342
// MI455X (gfx1250) — matrix operand order repaired and checked by reading, not compiled, not run on hardware
//
#include <hip/hip_runtime.h>

// ---------------- problem constants ----------------
#define BATCH   64
#define SYN     784
#define SPAD    800          // SYN padded to 25*32 (zero rows)
#define SBLKS   25           // 800/32 K-chunks of 32
#define TIME    500
#define TPAD    624          // padded time axis: 48 front pad + 500 + tail pad
#define KS      48           // kernel length (lags)
#define OCN     2000
#define NPAD    2048         // OCN padded to 8*256
#define TOUT    549          // conv output length: 500+96-48+1
#define THETA   40.0f
#define WINNERS 5
#define FODEP   48

#define TILE_M  64
#define N_TILE  256
#define NSTEPS  (KS * SBLKS)   // 1200 flattened (j, s_blk) K-steps
#define START_STEP 24          // j=0,sb=24: G slab is zero there; makes 1176 = 588 pairs
#define NPAIRS  ((NSTEPS - START_STEP) / 2)   // 588
#define STEP_BYTES ((size_t)NPAD * 64)        // 131072 B of G per K-step

// ws layout (bytes)
#define XT_BYTES ((size_t)BATCH * TPAD * SPAD * 2)                 //  63,897,600
#define G_BYTES  ((size_t)KS * SBLKS * NPAD * 32 * 2)              // 157,286,400
// pot after those: BATCH*TOUT*OCN*4 = 281,088,000

typedef __attribute__((ext_vector_type(8)))  _Float16 v8h;
typedef __attribute__((ext_vector_type(16))) _Float16 v16h;
typedef __attribute__((ext_vector_type(8)))  float    v8f;

__device__ inline v16h cat16(v8h lo, v8h hi) {
  return __builtin_shufflevector(lo, hi, 0,1,2,3,4,5,6,7,8,9,10,11,12,13,14,15);
}

// ---------- prep: x [B,1,784,500] f32 -> XT [B][TPAD][SPAD] f16 (time-major, padded) ----------
__global__ __launch_bounds__(256) void snn_prep_x(const float* __restrict__ x,
                                                  _Float16* __restrict__ XT) {
  size_t idx = (size_t)blockIdx.x * 256 + threadIdx.x;
  if (idx >= (size_t)BATCH * TPAD * SPAD) return;
  int s    = (int)(idx % SPAD);
  int tpad = (int)((idx / SPAD) % TPAD);
  int b    = (int)(idx / ((size_t)SPAD * TPAD));
  float v = 0.0f;
  if (s < SYN && tpad >= KS && tpad < KS + TIME)
    v = x[((size_t)b * SYN + s) * TIME + (tpad - KS)];
  XT[idx] = (_Float16)v;
}

// ---------- prep: weight [2000,784] -> G[j][s_blk][o(2048)][s_in(32)] f16 ----------
// g(w, j) = max(0, min(j/16, 1.5*w - j/32));  zero for o>=2000 or s>=784
__global__ __launch_bounds__(256) void snn_prep_g(const float* __restrict__ w,
                                                  _Float16* __restrict__ G) {
  size_t idx = (size_t)blockIdx.x * 256 + threadIdx.x;
  if (idx >= (size_t)KS * SBLKS * NPAD * 32) return;
  int s_in = (int)(idx & 31);
  int o    = (int)((idx >> 5) & (NPAD - 1));
  int rest = (int)(idx >> 16);          // rest = j*25 + s_blk  (2048*32 = 65536)
  int sb   = rest % SBLKS;
  int j    = rest / SBLKS;
  int s    = sb * 32 + s_in;
  float wv = (o < OCN && s < SYN) ? w[(size_t)o * SYN + s] : 0.0f;
  float g  = 1.5f * wv - (float)j * (1.0f / 32.0f);
  float sp = (float)j * (1.0f / 16.0f);
  g = fminf(sp, g);
  g = fmaxf(0.0f, g);
  G[idx] = (_Float16)g;
}

__global__ __launch_bounds__(256) void snn_zero_f32(float* __restrict__ p, size_t n) {
  size_t i = (size_t)blockIdx.x * 256 + threadIdx.x;
  if (i < n) p[i] = 0.0f;
}

// ---------- WMMA implicit-GEMM conv:  pot[b][t'][o] = sum_{j,s} XT[b][t'+47-j][s] * G[j][s][o] ----------
// Block = 256 thr = 8 waves; tile M=64 x N=256; wave = 1 m-subtile x 8 n-subtiles.
// B tiles staged to LDS with gfx1250 async copies, 2 K-steps per stage, double-buffered.
// A fragments prefetched from global (L2-resident) above the barrier.
__global__ __launch_bounds__(256) void snn_gemm(const _Float16* __restrict__ XT,
                                                const _Float16* __restrict__ G,
                                                float* __restrict__ pot) {
  __shared__ __align__(16) _Float16 Bs[2][2][N_TILE * 32];   // 2 bufs x 2 steps x 16 KB

  const int b  = blockIdx.z;
  const int t0 = blockIdx.x * TILE_M;
  const int o0 = blockIdx.y * N_TILE;
  const int tid  = threadIdx.x;
  const int wave = tid >> 5;
  const int lane = tid & 31;
  const int lnid = lane & 15;       // position within 16-lane half
  const int lh   = lane >> 4;       // lane-half selector
  const int mw   = wave & 3;        // m-subtile (0..3)
  const int ng   = wave >> 2;       // n-group (0..1), 8 subtiles each

  // A fragment: row t' = rowt; lane-half picks K {0-7,16-23} vs {8-15,24-31}
  const int rowt = t0 + mw * 16 + lnid;
  // pointer for step=START_STEP (j=0, sb=24): u = rowt + 47 - j, + sb*32
  const _Float16* Aptr =
      XT + ((size_t)b * TPAD + (size_t)(rowt + KS - 1)) * SPAD + START_STEP * 32 + lh * 8;
  int sbc = START_STEP;             // running s_blk counter

  // B staging: per K-step the block's tile is a contiguous 16 KB slab of G;
  // thread tid copies one 64-byte column per step with 4 async b128 transfers.
  const char* Gbase = (const char*)G + (size_t)o0 * 64 + (size_t)tid * 64;
  const uint32_t lds0 = (uint32_t)(uintptr_t)(&Bs[0][0][0]) + (uint32_t)tid * 64;

#define STAGE_PAIR(pair, bufsel)                                                              \
  {                                                                                           \
    const char* gp0 = Gbase + ((size_t)START_STEP + 2 * (size_t)(pair)) * STEP_BYTES;         \
    const char* gp1 = gp0 + STEP_BYTES;                                                       \
    uint32_t lp0 = lds0 + (uint32_t)(bufsel) * (2 * N_TILE * 64);                             \
    uint32_t lp1 = lp0 + (N_TILE * 64);                                                       \
    asm volatile("global_load_async_to_lds_b128 %0, %1, off"           :: "v"(lp0), "v"(gp0) : "memory"); \
    asm volatile("global_load_async_to_lds_b128 %0, %1, off offset:16" :: "v"(lp0), "v"(gp0) : "memory"); \
    asm volatile("global_load_async_to_lds_b128 %0, %1, off offset:32" :: "v"(lp0), "v"(gp0) : "memory"); \
    asm volatile("global_load_async_to_lds_b128 %0, %1, off offset:48" :: "v"(lp0), "v"(gp0) : "memory"); \
    asm volatile("global_load_async_to_lds_b128 %0, %1, off"           :: "v"(lp1), "v"(gp1) : "memory"); \
    asm volatile("global_load_async_to_lds_b128 %0, %1, off offset:16" :: "v"(lp1), "v"(gp1) : "memory"); \
    asm volatile("global_load_async_to_lds_b128 %0, %1, off offset:32" :: "v"(lp1), "v"(gp1) : "memory"); \
    asm volatile("global_load_async_to_lds_b128 %0, %1, off offset:48" :: "v"(lp1), "v"(gp1) : "memory"); \
  }

// advance A pointer by one K-step: sb++ (+32 elems); on sb wrap, j++ (-SPAD), sb->0
#define ADVANCE_A()                                                        \
  {                                                                        \
    if (++sbc == SBLKS) { sbc = 0; Aptr -= ((SBLKS - 1) * 32 + SPAD); }    \
    else                { Aptr += 32; }                                    \
  }

  STAGE_PAIR(0, 0)

  v8f acc[8] = {v8f{}, v8f{}, v8f{}, v8f{}, v8f{}, v8f{}, v8f{}, v8f{}};
  int buf = 0;

#pragma unroll 1
  for (int pair = 0; pair < NPAIRS; ++pair) {
    // prefetch both A fragments for this pair BEFORE the sync: they don't
    // touch LDS, so their global latency hides behind the barrier wait
    v16h Af[2];
    {
      v8h a0 = *(const v8h*)(Aptr);
      v8h a1 = *(const v8h*)(Aptr + 16);
      Af[0] = cat16(a0, a1);
      ADVANCE_A()
      v8h a2 = *(const v8h*)(Aptr);
      v8h a3 = *(const v8h*)(Aptr + 16);
      Af[1] = cat16(a2, a3);
      ADVANCE_A()
    }

    // my async fills done; barrier publishes buf's fill to all waves and
    // guarantees every wave finished computing from buf^1 (safe to overwrite)
    asm volatile("s_wait_asynccnt 0x0" ::: "memory");
    __syncthreads();

    if (pair + 1 < NPAIRS) STAGE_PAIR(pair + 1, buf ^ 1)

#pragma unroll
    for (int h = 0; h < 2; ++h) {      // the two K-steps staged in this buffer
      const _Float16* Bbase = &Bs[buf][h][(ng * 128 + lnid) * 32 + lh * 8];
      // software-pipeline B fragments one ahead so each WMMA overlaps the
      // next fragment's LDS round-trip (graduated s_wait_dscnt)
      v16h Bcur;
      {
        v8h b0 = *(const v8h*)(Bbase);
        v8h b1 = *(const v8h*)(Bbase + 16);
        Bcur = cat16(b0, b1);
      }
#pragma unroll
      for (int k = 0; k < 8; ++k) {
        v16h Bnext = Bcur;
        if (k < 7) {
          const _Float16* bp = Bbase + (k + 1) * (16 * 32);
          v8h b0 = *(const v8h*)(bp);
          v8h b1 = *(const v8h*)(bp + 16);
          Bnext = cat16(b0, b1);
        }
        acc[k] = __builtin_amdgcn_wmma_f32_16x16x32_f16(
            false, Af[h], false, Bcur, (short)0, acc[k], false, false);
        Bcur = Bnext;
      }
    }
    buf ^= 1;
  }

  // C/D layout: VGPR r, lanes 0-15: M=r, lanes 16-31: M=8+r; N = lane&15
#pragma unroll
  for (int k = 0; k < 8; ++k) {
#pragma unroll
    for (int r = 0; r < 8; ++r) {
      int t = t0 + mw * 16 + lh * 8 + r;
      int o = o0 + ng * 128 + k * 16 + lnid;
      if (t < TOUT && o < OCN)
        pot[((size_t)b * TOUT + t) * OCN + o] = acc[k][r];
    }
  }
#undef STAGE_PAIR
#undef ADVANCE_A
}

// ---------- sequential winner-takes-all over time, one block per batch ----------
__global__ __launch_bounds__(256) void snn_wta(const float* __restrict__ pot,
                                               const float* __restrict__ bias,
                                               float* __restrict__ out) {
  const int b   = blockIdx.x;
  const int tid = threadIdx.x;
  __shared__ int   dep[200];
  __shared__ float sbias[OCN];
  __shared__ float rv[256];
  __shared__ int   rf[256];
  __shared__ int   rc[256];

  for (int o = tid; o < OCN; o += 256) sbias[o] = THETA * bias[o];
  if (tid < 200) dep[tid] = 0;
  __syncthreads();

  for (int t = 0; t < TOUT; ++t) {
    const float* p = pot + ((size_t)b * TOUT + t) * OCN;
    float bv = 0.0f;
    int   bf = 1 << 30;
    for (int o = tid; o < OCN; o += 256) {
      int c = o / 200, n = o % 200;            // pot row o = c*200 + n
      float v = (dep[n] == 0) ? (p[o] + sbias[o]) : 0.0f;
      int f = n * 10 + c;                      // reference flat argmax order
      if (v > bv || (v == bv && f < bf)) { bv = v; bf = f; }
    }
    rv[tid] = bv;
    rf[tid] = bf;
    rc[tid] = (tid < 200 && dep[tid] != 0) ? 1 : 0;
    __syncthreads();
    for (int s = 128; s > 0; s >>= 1) {
      if (tid < s) {
        rc[tid] += rc[tid + s];
        float ov = rv[tid + s];
        int   of = rf[tid + s];
        if (ov > rv[tid] || (ov == rv[tid] && of < rf[tid])) {
          rv[tid] = ov; rf[tid] = of;
        }
      }
      __syncthreads();
    }
    float winv = rv[0];
    int   winf = rf[0];
    int   cnt  = rc[0];
    bool spike = (cnt < WINNERS) && (winv > THETA);
    if (spike && tid == 0) {
      int n = winf / 10, c = winf % 10;
      out[(((size_t)b * 10 + c) * 200 + n) * TOUT + t] = 1.0f;
      dep[n] += FODEP;                         // then global clip(dep-1,0,47) below
    }
    __syncthreads();
    if (tid < 200) {
      int d = dep[tid] - 1;
      dep[tid] = d < 0 ? 0 : (d > FODEP - 1 ? FODEP - 1 : d);
    }
    __syncthreads();
  }
}

// ---------------- host launch ----------------
extern "C" void kernel_launch(void* const* d_in, const int* in_sizes, int n_in,
                              void* d_out, int out_size, void* d_ws, size_t ws_size,
                              hipStream_t stream) {
  const float* x    = (const float*)d_in[0];   // [64,1,784,500]
  const float* w    = (const float*)d_in[1];   // [2000,784]
  const float* bias = (const float*)d_in[2];   // [2000]
  float* out = (float*)d_out;                  // [64,10,200,549]

  char* ws = (char*)d_ws;
  _Float16* XT = (_Float16*)ws;
  _Float16* G  = (_Float16*)(ws + XT_BYTES);
  float*    pt = (float*)(ws + XT_BYTES + G_BYTES);

  {
    size_t n = (size_t)BATCH * TPAD * SPAD;
    snn_prep_x<<<dim3((unsigned)((n + 255) / 256)), dim3(256), 0, stream>>>(x, XT);
  }
  {
    size_t n = (size_t)KS * SBLKS * NPAD * 32;
    snn_prep_g<<<dim3((unsigned)((n + 255) / 256)), dim3(256), 0, stream>>>(w, G);
  }
  {
    size_t n = (size_t)out_size;
    snn_zero_f32<<<dim3((unsigned)((n + 255) / 256)), dim3(256), 0, stream>>>(out, n);
  }
  snn_gemm<<<dim3((TOUT + TILE_M - 1) / TILE_M, NPAD / N_TILE, BATCH),
             dim3(256), 0, stream>>>(XT, G, pt);
  snn_wta<<<dim3(BATCH), dim3(256), 0, stream>>>(pt, bias, out);
}
